// Head_24644522344740
// MI455X (gfx1250) — matrix operand order repaired and checked by reading, not compiled, not run on hardware
//
#include <hip/hip_runtime.h>
#include <hip/hip_bf16.h>

typedef __attribute__((ext_vector_type(16))) __bf16 v16bf;
typedef __attribute__((ext_vector_type(8)))  __bf16 v8bf;
typedef __attribute__((ext_vector_type(8)))  float  v8f;

union V16 { v16bf v; v8bf h[2]; };

#define WMMA_BF16(A, B, C) \
  __builtin_amdgcn_wmma_f32_16x16x32_bf16(false, (A), false, (B), (short)0, (C), false, false)

namespace {
constexpr int Bn = 128, Tn = 256, Cn = 384, Hn = 64;
constexpr int WT_ELEMS = Hn * Cn;                 // 24576 bf16 per weight matrix
constexpr int Q_OFF    = 3 * WT_ELEMS;            // q  [T][H] bf16
constexpr int K_OFF    = Q_OFF + Tn * Hn;         // k  [T][H] bf16
constexpr int VT_OFF   = K_OFF + Tn * Hn;         // vT [H][T] bf16
constexpr int SMEM_ELEMS = VT_OFF + Hn * Tn;      // 122880 elems
constexpr int SMEM_BYTES = SMEM_ELEMS * 2;        // 245760 B < 320KB/WGP
constexpr float SCALE = 0.05103103630798287f;     // 384^-0.5 (ref scales by C, not H)
}

__global__ void __launch_bounds__(256)
head_attn_kernel(const float* __restrict__ x, const float* __restrict__ Wq,
                 const float* __restrict__ Wk, const float* __restrict__ Wv,
                 float* __restrict__ out)
{
  extern __shared__ __bf16 sm[];
  const int b    = blockIdx.x;
  const int tid  = threadIdx.x;
  const int wave = tid >> 5;
  const int lane = tid & 31;
  const int hl   = lane >> 4;   // half-wave (lanes 16-31)
  const int ln   = lane & 15;

  // ---- Phase 0: stage W^T (bf16, [H][C] row-major) into LDS ----
  {
    const float* Ws[3] = {Wq, Wk, Wv};
    for (int mat = 0; mat < 3; ++mat) {
      const float* W = Ws[mat];
      __bf16* wt = sm + mat * WT_ELEMS;
      for (int idx = tid; idx < Cn * Hn; idx += 256) {
        const int kc = idx >> 6;     // C index
        const int n  = idx & 63;     // H index
        wt[n * Cn + kc] = (__bf16)W[idx];
      }
    }
  }
  __syncthreads();

  // ---- Phase 1: q/k/v = x @ W, bf16 results in LDS ----
  const float* xb = x + (size_t)b * Tn * Cn;
  for (int mti = 0; mti < 2; ++mti) {
    const int t0 = (wave + mti * 8) * 16;
    v8f acc[3][4];
    #pragma unroll
    for (int m = 0; m < 3; ++m)
      #pragma unroll
      for (int nt = 0; nt < 4; ++nt)
        #pragma unroll
        for (int i = 0; i < 8; ++i) acc[m][nt][i] = 0.0f;

    const float* arow = xb + (t0 + ln) * Cn + hl * 16;
    #pragma unroll
    for (int kk = 0; kk < 12; ++kk) {
      const float4 f0 = *(const float4*)(arow + kk * 32 + 0);
      const float4 f1 = *(const float4*)(arow + kk * 32 + 4);
      const float4 g0 = *(const float4*)(arow + kk * 32 + 8);
      const float4 g1 = *(const float4*)(arow + kk * 32 + 12);
      v16bf a;
      a[0]=(__bf16)f0.x;  a[1]=(__bf16)f0.y;  a[2]=(__bf16)f0.z;  a[3]=(__bf16)f0.w;
      a[4]=(__bf16)f1.x;  a[5]=(__bf16)f1.y;  a[6]=(__bf16)f1.z;  a[7]=(__bf16)f1.w;
      a[8]=(__bf16)g0.x;  a[9]=(__bf16)g0.y;  a[10]=(__bf16)g0.z; a[11]=(__bf16)g0.w;
      a[12]=(__bf16)g1.x; a[13]=(__bf16)g1.y; a[14]=(__bf16)g1.z; a[15]=(__bf16)g1.w;
      #pragma unroll
      for (int mat = 0; mat < 3; ++mat) {
        #pragma unroll
        for (int nt = 0; nt < 4; ++nt) {
          const v16bf bw = *(const v16bf*)(sm + mat * WT_ELEMS +
                                           (nt * 16 + ln) * Cn + kk * 32 + hl * 16);
          acc[mat][nt] = WMMA_BF16(a, bw, acc[mat][nt]);
        }
      }
    }
    #pragma unroll
    for (int nt = 0; nt < 4; ++nt) {
      const int h = nt * 16 + ln;
      #pragma unroll
      for (int i = 0; i < 8; ++i) {
        const int t = t0 + i + 8 * hl;
        sm[Q_OFF  + t * Hn + h] = (__bf16)acc[0][nt][i];
        sm[K_OFF  + t * Hn + h] = (__bf16)acc[1][nt][i];
        sm[VT_OFF + h * Tn + t] = (__bf16)acc[2][nt][i];   // v stored transposed
      }
    }
  }
  __syncthreads();

  // ---- Phase 2: causal flash attention ----
  __bf16* Pb = sm + wave * 512;   // wave-private 16x32 bf16 bounce (overlays dead Wt)
  for (int mti = 0; mti < 2; ++mti) {
    const int mt = mti ? (15 - wave) : wave;  // balance causal triangle
    const int t0 = mt * 16;

    const __bf16* krow = sm + K_OFF + (t0 + ln) * Hn + hl * 16;
    V16 ak0, ak1;
    ak0.h[0] = *(const v8bf*)(krow + 0);
    ak0.h[1] = *(const v8bf*)(krow + 8);
    ak1.h[0] = *(const v8bf*)(krow + 32);
    ak1.h[1] = *(const v8bf*)(krow + 40);

    v8f oacc[4];
    float rmax[8], rsum[8];
    #pragma unroll
    for (int nt = 0; nt < 4; ++nt)
      #pragma unroll
      for (int i = 0; i < 8; ++i) oacc[nt][i] = 0.0f;
    #pragma unroll
    for (int i = 0; i < 8; ++i) { rmax[i] = -3.0e38f; rsum[i] = 0.0f; }

    for (int sb = 0; sb <= t0 + 15; sb += 32) {
      v8f S0, S1;
      #pragma unroll
      for (int i = 0; i < 8; ++i) { S0[i] = 0.0f; S1[i] = 0.0f; }
      {
        const __bf16* q0 = sm + Q_OFF + (sb + ln) * Hn + hl * 16;
        const __bf16* q1 = sm + Q_OFF + (sb + 16 + ln) * Hn + hl * 16;
        const v16bf bq00 = *(const v16bf*)(q0);
        const v16bf bq01 = *(const v16bf*)(q0 + 32);
        const v16bf bq10 = *(const v16bf*)(q1);
        const v16bf bq11 = *(const v16bf*)(q1 + 32);
        S0 = WMMA_BF16(ak0.v, bq00, S0);
        S0 = WMMA_BF16(ak1.v, bq01, S0);
        S1 = WMMA_BF16(ak0.v, bq10, S1);
        S1 = WMMA_BF16(ak1.v, bq11, S1);
      }

      float mnew[8];
      const int s0c = sb + ln, s1c = sb + 16 + ln;
      #pragma unroll
      for (int i = 0; i < 8; ++i) {
        const int t = t0 + i + 8 * hl;
        const float a0 = (s0c <= t) ? S0[i] * SCALE : -1.0e30f;
        const float a1 = (s1c <= t) ? S1[i] * SCALE : -1.0e30f;
        S0[i] = a0; S1[i] = a1;
        float m = fmaxf(a0, a1);                      // row lives in one 16-lane half
        m = fmaxf(m, __shfl_xor(m, 1, 32));
        m = fmaxf(m, __shfl_xor(m, 2, 32));
        m = fmaxf(m, __shfl_xor(m, 4, 32));
        m = fmaxf(m, __shfl_xor(m, 8, 32));
        mnew[i] = fmaxf(rmax[i], m);
      }
      #pragma unroll
      for (int i = 0; i < 8; ++i) {
        const float corr = __expf(rmax[i] - mnew[i]);
        rmax[i] = mnew[i];
        const float p0 = __expf(S0[i] - mnew[i]);
        const float p1 = __expf(S1[i] - mnew[i]);
        float rs = p0 + p1;
        rs += __shfl_xor(rs, 1, 32);
        rs += __shfl_xor(rs, 2, 32);
        rs += __shfl_xor(rs, 4, 32);
        rs += __shfl_xor(rs, 8, 32);
        rsum[i] = rsum[i] * corr + rs;
        #pragma unroll
        for (int nt = 0; nt < 4; ++nt) oacc[nt][i] *= corr;
        Pb[(i + 8 * hl) * 32 + ln]      = (__bf16)p0;   // C-layout -> row-major P
        Pb[(i + 8 * hl) * 32 + 16 + ln] = (__bf16)p1;
      }

      V16 ap;                                           // row-major P -> A-layout
      ap.h[0] = *(const v8bf*)(Pb + ln * 32 + hl * 16);
      ap.h[1] = *(const v8bf*)(Pb + ln * 32 + hl * 16 + 8);
      #pragma unroll
      for (int nt = 0; nt < 4; ++nt) {
        const v16bf bv = *(const v16bf*)(sm + VT_OFF + (nt * 16 + ln) * Tn + sb + hl * 16);
        oacc[nt] = WMMA_BF16(ap.v, bv, oacc[nt]);
      }
    }

    float rinv[8];
    #pragma unroll
    for (int i = 0; i < 8; ++i) rinv[i] = 1.0f / rsum[i];
    float* ob = out + ((size_t)b * Tn + t0) * Hn;
    #pragma unroll
    for (int nt = 0; nt < 4; ++nt) {
      #pragma unroll
      for (int i = 0; i < 8; ++i) {
        ob[(i + 8 * hl) * Hn + nt * 16 + ln] = oacc[nt][i] * rinv[i];
      }
    }
  }
}

extern "C" void kernel_launch(void* const* d_in, const int* in_sizes, int n_in,
                              void* d_out, int out_size, void* d_ws, size_t ws_size,
                              hipStream_t stream) {
  (void)in_sizes; (void)n_in; (void)d_ws; (void)ws_size; (void)out_size;
  const float* x  = (const float*)d_in[0];
  const float* Wq = (const float*)d_in[1];
  const float* Wk = (const float*)d_in[2];
  const float* Wv = (const float*)d_in[3];
  float* out = (float*)d_out;
  hipFuncSetAttribute((const void*)head_attn_kernel,
                      hipFuncAttributeMaxDynamicSharedMemorySize, SMEM_BYTES);
  head_attn_kernel<<<Bn, 256, SMEM_BYTES, stream>>>(x, Wq, Wk, Wv, out);
}
